// ProbLSTMCell_85194971283616
// MI455X (gfx1250) — compile-verified, matrix operand order checked by reading, not run on hardware
//
#include <hip/hip_runtime.h>
#include <hip/hip_bf16.h>

// CDNA5 / gfx1250: wave32, WMMA 16x16x32 bf16, async global->LDS staging.
typedef __attribute__((ext_vector_type(16))) __bf16 bf16x16;
typedef __attribute__((ext_vector_type(8)))  float  f32x8;
typedef __attribute__((__vector_size__(16))) int    v4i;

#define BDIM 8192
#define FDIM 512
#define HDIM 1024
#define GDIM 4096   // 4*H

#define M_TILE 128      // batch rows per block (8 waves x 16 rows)
#define H_TILE 32       // h-columns per block (x4 gates = 128 weight rows staged)
#define KSTEP  32       // WMMA K per step
#define LDS_LD 40       // halves per staged row: 32 data + 8 pad (80B row stride)
#define NSTEP  ((FDIM + HDIM) / KSTEP)   // 48 k-steps total
#define PH0    (FDIM / KSTEP)            // 16 steps in phase 0

// bf16 workspace layout (ushort elements)
#define WSE_INP 0
#define WSE_HX  (BDIM * FDIM)
#define WSE_WIH (WSE_HX + BDIM * HDIM)
#define WSE_WHH (WSE_WIH + GDIM * FDIM)
#define WSE_TOT (WSE_WHH + GDIM * HDIM)  // 18,874,368 elems = 37,748,736 B

// ---------------- small helpers ----------------

__device__ __forceinline__ unsigned int pk_bf16(float x, float y) {
#if __has_builtin(__builtin_amdgcn_cvt_pk_bf16_f32)
  typedef __attribute__((ext_vector_type(2))) __bf16 bf16x2;
  bf16x2 v = __builtin_amdgcn_cvt_pk_bf16_f32(x, y);
  return __builtin_bit_cast(unsigned int, v);
#else
  unsigned int ux = __float_as_uint(x), uy = __float_as_uint(y);
  ux = (ux + 0x7FFFu + ((ux >> 16) & 1u)) >> 16;
  uy = (uy + 0x7FFFu + ((uy >> 16) & 1u)) >> 16;
  return ux | (uy << 16);
#endif
}

__device__ __forceinline__ float fast_tanh(float x) {
#if __has_builtin(__builtin_amdgcn_tanhf)
  return __builtin_amdgcn_tanhf(x);          // v_tanh_f32 (gfx1250 trans op)
#elif __has_builtin(__builtin_amdgcn_tanh_f32)
  return __builtin_amdgcn_tanh_f32(x);
#else
  return tanhf(x);
#endif
}
__device__ __forceinline__ float fast_sigmoid(float x) {
  return 0.5f * fast_tanh(0.5f * x) + 0.5f;
}

__device__ __forceinline__ void wait_async0() {
#if __has_builtin(__builtin_amdgcn_s_wait_asynccnt)
  __builtin_amdgcn_s_wait_asynccnt(0);
#else
  asm volatile("s_wait_asynccnt 0" ::: "memory");
#endif
}

// 16B per-lane copy global(bf16) -> LDS; async engine if available.
// Builtin prototype (from hipcc diagnostic): (v4i AS1*, v4i AS3*, imm, imm).
__device__ __forceinline__ void async_cp16(const unsigned short* g,
                                           unsigned short* l) {
#if __has_builtin(__builtin_amdgcn_global_load_async_to_lds_b128)
  __builtin_amdgcn_global_load_async_to_lds_b128(
      (__attribute__((address_space(1))) v4i*)
          (__attribute__((address_space(1))) void*)const_cast<unsigned short*>(g),
      (__attribute__((address_space(3))) v4i*)
          (__attribute__((address_space(3))) void*)l,
      0, 0);
#else
  *reinterpret_cast<uint4*>(l) = *reinterpret_cast<const uint4*>(g);
#endif
}

// Read one 16x32 bf16 fragment row for this lane from LDS.
// 16-bit A/B layout: lane holds K=[kbase,kbase+8) in v[0..3] and
// K=[16+kbase,16+kbase+8) in v[4..7]; kbase = 0 (lanes 0-15) / 8 (16-31).
__device__ __forceinline__ bf16x16 read_frag(const unsigned short* row, int kbase) {
  union { uint4 q[2]; bf16x16 v; } u;
  u.q[0] = *reinterpret_cast<const uint4*>(row + kbase);
  u.q[1] = *reinterpret_cast<const uint4*>(row + 16 + kbase);
  return u.v;
}

// One k-step: hoist all ds reads, then burst 8 WMMAs. acc[t], t = q*2+s.
__device__ __forceinline__ void compute_tile(const unsigned short* tA,
                                             const unsigned short* tB,
                                             f32x8 acc[8], int wave, int lane) {
  const int rsel = lane & 15;
  const int kbase = (lane >> 4) << 3;
  bf16x16 af = read_frag(tA + (wave * 16 + rsel) * LDS_LD, kbase);
  bf16x16 bf[8];
#pragma unroll
  for (int t = 0; t < 8; ++t) {
    const int q = t >> 1, s = t & 1;
    bf[t] = read_frag(tB + (q * 32 + s * 16 + rsel) * LDS_LD, kbase);
  }
#pragma unroll
  for (int t = 0; t < 8; ++t)
    acc[t] = __builtin_amdgcn_wmma_f32_16x16x32_bf16(
        false, af, false, bf[t], (short)0, acc[t], false, false);
}

// Fused LSTM epilogue. C/D layout: VGPR v -> M = v + 8*(lane>=16); N = lane&15.
__device__ __forceinline__ void lstm_epilogue(
    f32x8 acc[8], const float* __restrict__ bih, const float* __restrict__ bhh,
    const float* __restrict__ mu, const float* __restrict__ sig,
    const float* __restrict__ eps, const float* __restrict__ cx,
    float* __restrict__ out, int m0, int h0, int wave, int lane) {
  const int rsel = lane & 15;
  float badd[4][2], bsig[4][2];
#pragma unroll
  for (int q = 0; q < 4; ++q)
#pragma unroll
    for (int s = 0; s < 2; ++s) {
      int col = q * HDIM + h0 + s * 16 + rsel;
      badd[q][s] = bih[col] + bhh[col] + mu[col];
      bsig[q][s] = sig[col];
    }
  const int mrow0 = m0 + wave * 16 + ((lane >> 4) << 3);
#pragma unroll
  for (int v = 0; v < 8; ++v) {
    const int row = mrow0 + v;
    const float* erow = eps + (size_t)row * GDIM;
#pragma unroll
    for (int s = 0; s < 2; ++s) {
      const int h = h0 + s * 16 + rsel;
      float gi = acc[0 * 2 + s][v] + badd[0][s] + bsig[0][s] * erow[0 * HDIM + h];
      float gf = acc[1 * 2 + s][v] + badd[1][s] + bsig[1][s] * erow[1 * HDIM + h];
      float gg = acc[2 * 2 + s][v] + badd[2][s] + bsig[2][s] * erow[2 * HDIM + h];
      float go = acc[3 * 2 + s][v] + badd[3][s] + bsig[3][s] * erow[3 * HDIM + h];
      float c  = cx[(size_t)row * HDIM + h];
      float cy = fast_sigmoid(gf) * c + fast_sigmoid(gi) * fast_tanh(gg);
      float hy = fast_sigmoid(go) * fast_tanh(cy);
      out[(size_t)row * HDIM + h] = hy;                        // hy
      out[(size_t)BDIM * HDIM + (size_t)row * HDIM + h] = cy;  // cy
    }
  }
}

// ---------------- prep: fp32 -> bf16 pass (read 75MB, write 37.7MB) ----------------
extern "C" __global__ __launch_bounds__(256)
void cvt_f32_to_bf16(const float* __restrict__ src,
                     unsigned short* __restrict__ dst, int n) {
  int i = (blockIdx.x * blockDim.x + threadIdx.x) * 8;
  if (i >= n) return;
  const float4* s = reinterpret_cast<const float4*>(src + i);
  float4 a = s[0], b = s[1];
  uint4 o = make_uint4(pk_bf16(a.x, a.y), pk_bf16(a.z, a.w),
                       pk_bf16(b.x, b.y), pk_bf16(b.z, b.w));
  *reinterpret_cast<uint4*>(dst + i) = o;
}

// ---------------- main: async-staged bf16 WMMA GEMM + fused LSTM ----------------
extern "C" __global__ __launch_bounds__(256)
void prob_lstm_wmma_async(const unsigned short* __restrict__ wsb,
                          const float* __restrict__ cx,
                          const float* __restrict__ bih, const float* __restrict__ bhh,
                          const float* __restrict__ mu,  const float* __restrict__ sig,
                          const float* __restrict__ eps, float* __restrict__ out) {
  __shared__ __align__(16) unsigned short sA[2][M_TILE * LDS_LD];  // 2 x 10KB
  __shared__ __align__(16) unsigned short sB[2][M_TILE * LDS_LD];  // 2 x 10KB

  const int tid  = threadIdx.x;
  const int wave = tid >> 5;
  const int lane = tid & 31;
  const int m0 = blockIdx.x * M_TILE;
  const int h0 = blockIdx.y * H_TILE;

  const unsigned short* inpb = wsb + WSE_INP;
  const unsigned short* hxb  = wsb + WSE_HX;
  const unsigned short* wihb = wsb + WSE_WIH;
  const unsigned short* whhb = wsb + WSE_WHH;

  f32x8 acc[8];
#pragma unroll
  for (int t = 0; t < 8; ++t) { f32x8 z = {}; acc[t] = z; }

  // Issue one 128x32 A tile + 128x32 B tile as 4 x b128 async copies/thread.
  auto issue = [&](int buf, int t) {
    const int phase = (t >= PH0);
    const int k0 = (t - (phase ? PH0 : 0)) * KSTEP;
    const unsigned short* ab = phase ? hxb  : inpb;
    const unsigned short* wb = phase ? whhb : wihb;
    const int K = phase ? HDIM : FDIM;
#pragma unroll
    for (int j = 0; j < 2; ++j) {
      const int c = tid + j * 256;      // 512 x 16B chunks per tile
      const int row = c >> 2, qq = c & 3;
      async_cp16(ab + (size_t)(m0 + row) * K + k0 + qq * 8,
                 &sA[buf][row * LDS_LD + qq * 8]);
      const int wrow = (row >> 5) * HDIM + h0 + (row & 31);
      async_cp16(wb + (size_t)wrow * K + k0 + qq * 8,
                 &sB[buf][row * LDS_LD + qq * 8]);
    }
  };

  issue(0, 0);
  wait_async0();
  __syncthreads();
  for (int t = 0; t < NSTEP; ++t) {
    if (t + 1 < NSTEP) issue((t + 1) & 1, t + 1);  // async copy overlaps WMMA
    compute_tile(sA[t & 1], sB[t & 1], acc, wave, lane);
    wait_async0();
    __syncthreads();
  }

  lstm_epilogue(acc, bih, bhh, mu, sig, eps, cx, out, m0, h0, wave, lane);
}

// ---------------- fallback: fp32-direct staging (no workspace needed) ----------------
__device__ __forceinline__ void cvt_store16(const float* __restrict__ rp,
                                            unsigned short* dst) {
  const float4* s = reinterpret_cast<const float4*>(rp);
  float4 a = s[0], b = s[1], c = s[2], d = s[3];
  uint4 q0 = make_uint4(pk_bf16(a.x, a.y), pk_bf16(a.z, a.w),
                        pk_bf16(b.x, b.y), pk_bf16(b.z, b.w));
  uint4 q1 = make_uint4(pk_bf16(c.x, c.y), pk_bf16(c.z, c.w),
                        pk_bf16(d.x, d.y), pk_bf16(d.z, d.w));
  uint4* o = reinterpret_cast<uint4*>(dst);
  o[0] = q0; o[1] = q1;
}

extern "C" __global__ __launch_bounds__(256)
void prob_lstm_wmma_f32(const float* __restrict__ inp, const float* __restrict__ hx,
                        const float* __restrict__ cx,
                        const float* __restrict__ wih, const float* __restrict__ whh,
                        const float* __restrict__ bih, const float* __restrict__ bhh,
                        const float* __restrict__ mu,  const float* __restrict__ sig,
                        const float* __restrict__ eps, float* __restrict__ out) {
  __shared__ __align__(16) unsigned short sA[M_TILE * LDS_LD];
  __shared__ __align__(16) unsigned short sB[M_TILE * LDS_LD];

  const int tid = threadIdx.x, wave = tid >> 5, lane = tid & 31;
  const int m0 = blockIdx.x * M_TILE, h0 = blockIdx.y * H_TILE;
  const int srow = tid >> 1, shalf = tid & 1;
  const int bgate = srow >> 5, bj = srow & 31;

  f32x8 acc[8];
#pragma unroll
  for (int t = 0; t < 8; ++t) { f32x8 z = {}; acc[t] = z; }

  for (int phase = 0; phase < 2; ++phase) {
    const float* A = phase ? hx  : inp;
    const float* W = phase ? whh : wih;
    const int    K = phase ? HDIM : FDIM;
    const float* arow = A + (size_t)(m0 + srow) * K + shalf * 16;
    const float* brow = W + (size_t)(bgate * HDIM + h0 + bj) * K + shalf * 16;
    for (int k0 = 0; k0 < K; k0 += KSTEP) {
      __syncthreads();
      cvt_store16(arow + k0, &sA[srow * LDS_LD + shalf * 16]);
      cvt_store16(brow + k0, &sB[srow * LDS_LD + shalf * 16]);
      if (k0 + KSTEP < K) {
        __builtin_prefetch(arow + k0 + KSTEP, 0, 0);
        __builtin_prefetch(brow + k0 + KSTEP, 0, 0);
      }
      __syncthreads();
      compute_tile(sA, sB, acc, wave, lane);
    }
  }
  lstm_epilogue(acc, bih, bhh, mu, sig, eps, cx, out, m0, h0, wave, lane);
}

// ---------------- launch ----------------
extern "C" void kernel_launch(void* const* d_in, const int* in_sizes, int n_in,
                              void* d_out, int out_size, void* d_ws, size_t ws_size,
                              hipStream_t stream) {
  const float* inp = (const float*)d_in[0];
  const float* hx  = (const float*)d_in[1];
  const float* cx  = (const float*)d_in[2];
  const float* wih = (const float*)d_in[3];
  const float* whh = (const float*)d_in[4];
  const float* bih = (const float*)d_in[5];
  const float* bhh = (const float*)d_in[6];
  const float* mu  = (const float*)d_in[7];
  const float* sig = (const float*)d_in[8];
  const float* eps = (const float*)d_in[9];
  float* out = (float*)d_out;

  dim3 grid(BDIM / M_TILE, HDIM / H_TILE);  // 64 x 32

  if (ws_size >= (size_t)WSE_TOT * sizeof(unsigned short)) {
    unsigned short* wsb = (unsigned short*)d_ws;
    struct { const float* s; unsigned short* d; int n; } cv[4] = {
        {inp, wsb + WSE_INP, BDIM * FDIM},
        {hx,  wsb + WSE_HX,  BDIM * HDIM},
        {wih, wsb + WSE_WIH, GDIM * FDIM},
        {whh, wsb + WSE_WHH, GDIM * HDIM}};
    for (int i = 0; i < 4; ++i) {
      int blocks = (cv[i].n / 8 + 255) / 256;
      cvt_f32_to_bf16<<<blocks, 256, 0, stream>>>(cv[i].s, cv[i].d, cv[i].n);
    }
    prob_lstm_wmma_async<<<grid, 256, 0, stream>>>(wsb, cx, bih, bhh, mu, sig,
                                                   eps, out);
  } else {
    prob_lstm_wmma_f32<<<grid, 256, 0, stream>>>(inp, hx, cx, wih, whh, bih,
                                                 bhh, mu, sig, eps, out);
  }
}
